// Attention_85452669321838
// MI455X (gfx1250) — matrix operand order repaired and checked by reading, not compiled, not run on hardware
//
#include <hip/hip_runtime.h>

typedef __bf16 bf16_t;
typedef __attribute__((ext_vector_type(16))) __bf16 v16bf;
typedef __attribute__((ext_vector_type(8)))  __bf16 v8bf;
typedef __attribute__((ext_vector_type(8)))  float  v8f;
typedef __attribute__((ext_vector_type(4)))  int    v4i;

// pointer-to-AS1/AS3 v4i, as required by the async-load builtin prototype
typedef __attribute__((address_space(1))) v4i* global_v4i_p;
typedef __attribute__((address_space(3))) v4i* lds_v4i_p;

#define TILE_M 128
#define TILE_N 128
#define TILE_K 32
#define LDT    48   // LDS row stride in bf16 elements (96B -> every frag load 16B aligned)
#define LDTW   24   // same stride in dwords

#if defined(__gfx1250__) && __has_builtin(__builtin_amdgcn_global_load_async_to_lds_b128) && \
    __has_builtin(__builtin_amdgcn_s_wait_asynccnt)
#define USE_ASYNC 1
#else
#define USE_ASYNC 0
#endif

static __device__ __forceinline__ unsigned short bf16_bits(float x) {
  bf16_t h = (bf16_t)x;
  return __builtin_bit_cast(unsigned short, h);
}

// ---- WMMA fragment loaders (bf16, 16x16x32, NT layout) -------------------
static __device__ __forceinline__ v16bf frag_a(const bf16_t* tile, int row0, int lane) {
  const int mr = lane & 15, half = lane >> 4;
  const bf16_t* p = tile + (row0 + mr) * LDT + half * 8;
  v8bf lo = *(const v8bf*)(p);
  v8bf hi = *(const v8bf*)(p + 16);
  return __builtin_shufflevector(lo, hi, 0,1,2,3,4,5,6,7,8,9,10,11,12,13,14,15);
}
static __device__ __forceinline__ v16bf frag_b(const bf16_t* tile, int col0, int lane) {
  const int nr = lane & 15, half = lane >> 4;
  const bf16_t* p = tile + (col0 + nr) * LDT + half * 8;
  v8bf lo = *(const v8bf*)(p);
  v8bf hi = *(const v8bf*)(p + 16);
  return __builtin_shufflevector(lo, hi, 0,1,2,3,4,5,6,7,8,9,10,11,12,13,14,15);
}

// one 128x128x32 MMA step for the whole block (8 waves, each 64x32)
static __device__ __forceinline__ void mma_step(const bf16_t* As, const bf16_t* Bs,
                                                int wm, int wn, int lane, v8f acc[4][2]) {
  v16bf a[4], b[2];
#pragma unroll
  for (int i = 0; i < 4; ++i) a[i] = frag_a(As, wm + i * 16, lane);
#pragma unroll
  for (int j = 0; j < 2; ++j) b[j] = frag_b(Bs, wn + j * 16, lane);
#pragma unroll
  for (int i = 0; i < 4; ++i)
#pragma unroll
    for (int j = 0; j < 2; ++j)
      acc[i][j] = __builtin_amdgcn_wmma_f32_16x16x32_bf16(
          false, a[i], false, b[j], (short)0, acc[i][j], false, false);
}

// ---- tile movers ---------------------------------------------------------
// fp32 global -> bf16 LDS (convert through VGPRs); 128x32 tile, 256 threads
static __device__ __forceinline__ void ldg_tile_f32_bf16(const float* __restrict__ src,
                                                         bf16_t* dst, int row0, int ld,
                                                         int k0, int tid) {
  unsigned int* d = (unsigned int*)dst;
#pragma unroll
  for (int i = 0; i < 8; ++i) {
    int f = tid + 256 * i;
    int r = f >> 4;
    int c = f & 15;
    const float* s = src + (size_t)(row0 + r) * ld + k0 + c * 2;
    unsigned int u = (unsigned int)bf16_bits(s[0]) | ((unsigned int)bf16_bits(s[1]) << 16);
    d[r * LDTW + c] = u;
  }
}

#if USE_ASYNC
// bf16 global -> LDS via ASYNCcnt-tracked DMA; 2 x b128 per thread per tile
static __device__ __forceinline__ void issue_tile_async(const unsigned int* __restrict__ src,
                                                        bf16_t* dst, int row0, int ldw,
                                                        int k0w, int tid) {
#pragma unroll
  for (int i = 0; i < 2; ++i) {
    int f = tid + 256 * i;          // 512 16B-chunks per 128x32 bf16 tile
    int r = f >> 2;                 // 4 chunks per row
    int c = f & 3;
    const unsigned int* g = src + (size_t)(row0 + r) * ldw + k0w + c * 4;
    bf16_t* l = dst + r * LDT + c * 8;
    __builtin_amdgcn_global_load_async_to_lds_b128(
        (global_v4i_p)(g), (lds_v4i_p)(l), 0, 0);
  }
}
#else
// synchronous fallback
static __device__ __forceinline__ void ldg_tile_bf16(const unsigned int* __restrict__ src,
                                                     bf16_t* dst, int row0, int ldw,
                                                     int k0w, int tid) {
  unsigned int* d = (unsigned int*)dst;
#pragma unroll
  for (int i = 0; i < 8; ++i) {
    int f = tid + 256 * i;
    int r = f >> 4;
    int c = f & 15;
    d[r * LDTW + c] = src[(size_t)(row0 + r) * ldw + k0w + c];
  }
}
#endif

// double-buffered bf16-GEMM mainloop; nk MUST be even (32 or 128 here).
// Unrolled by 2 so each half-iteration uses statically-known LDS buffers.
static __device__ __forceinline__ void bf16_gemm_loop(
    const unsigned int* __restrict__ Aw, const unsigned int* __restrict__ Bw,
    bf16_t* As0, bf16_t* Bs0, bf16_t* As1, bf16_t* Bs1,
    int m0, int n0, int ldw, int nk, int wm, int wn, int lane, int tid, v8f acc[4][2]) {
#if USE_ASYNC
  issue_tile_async(Aw, As0, m0, ldw, 0, tid);
  issue_tile_async(Bw, Bs0, n0, ldw, 0, tid);
  for (int kk = 0; kk < nk; kk += 2) {
    // even step: prefetch tile kk+1 -> buf1, compute tile kk from buf0
    issue_tile_async(Aw, As1, m0, ldw, (kk + 1) * (TILE_K / 2), tid);
    issue_tile_async(Bw, Bs1, n0, ldw, (kk + 1) * (TILE_K / 2), tid);
    __builtin_amdgcn_s_wait_asynccnt(4);     // tile kk landed; kk+1 in flight
    __syncthreads();
    mma_step(As0, Bs0, wm, wn, lane, acc);
    __syncthreads();
    // odd step: prefetch tile kk+2 -> buf0 (unless last pair), compute tile kk+1
    if (kk + 2 < nk) {
      issue_tile_async(Aw, As0, m0, ldw, (kk + 2) * (TILE_K / 2), tid);
      issue_tile_async(Bw, Bs0, n0, ldw, (kk + 2) * (TILE_K / 2), tid);
      __builtin_amdgcn_s_wait_asynccnt(4);
    } else {
      __builtin_amdgcn_s_wait_asynccnt(0);
    }
    __syncthreads();
    mma_step(As1, Bs1, wm, wn, lane, acc);
    __syncthreads();
  }
#else
  (void)As1; (void)Bs1;
  for (int kk = 0; kk < nk; ++kk) {
    ldg_tile_bf16(Aw, As0, m0, ldw, kk * (TILE_K / 2), tid);
    ldg_tile_bf16(Bw, Bs0, n0, ldw, kk * (TILE_K / 2), tid);
    __syncthreads();
    mma_step(As0, Bs0, wm, wn, lane, acc);
    __syncthreads();
  }
#endif
}

#define GEMM_PREAMBLE()                                                \
  const int tid = threadIdx.x, lane = tid & 31, wid = tid >> 5;        \
  const int wm = (wid >> 2) * 64, wn = (wid & 3) * 32;                 \
  const int m0 = blockIdx.y * TILE_M, n0 = blockIdx.x * TILE_N;        \
  v8f acc[4][2];                                                       \
  _Pragma("unroll") for (int i = 0; i < 4; ++i)                        \
  _Pragma("unroll") for (int j = 0; j < 2; ++j) {                      \
    v8f z = {0.f, 0.f, 0.f, 0.f, 0.f, 0.f, 0.f, 0.f};                  \
    acc[i][j] = z;                                                     \
  }

// ---- kernel 0: zero the softmax-denominator accumulator ------------------
__global__ void init_zero_kernel(float* __restrict__ p) {
  p[blockIdx.x * 256 + threadIdx.x] = 0.0f;
}

// ---- kernel 1: Y = X @ W^T  (fp32 in, bf16 out; TRANS picks store layout)
template <int TRANS>
__global__ __launch_bounds__(256) void qkv_proj_kernel(
    const float* __restrict__ X, const float* __restrict__ W,
    bf16_t* __restrict__ Y, int kdim, int nrows, int ocols)
{
  __shared__ alignas(16) bf16_t As[TILE_M * LDT];
  __shared__ alignas(16) bf16_t Bs[TILE_N * LDT];
  GEMM_PREAMBLE();

  for (int k0 = 0; k0 < kdim; k0 += TILE_K) {
    ldg_tile_f32_bf16(X, As, m0, kdim, k0, tid);
    ldg_tile_f32_bf16(W, Bs, n0, kdim, k0, tid);
    __syncthreads();
    mma_step(As, Bs, wm, wn, lane, acc);
    __syncthreads();
  }

  const int nc = lane & 15, mb = (lane >> 4) * 8;
#pragma unroll
  for (int i = 0; i < 4; ++i)
#pragma unroll
    for (int j = 0; j < 2; ++j) {
      const int mrow = m0 + wm + i * 16 + mb;
      const int ncol = n0 + wn + j * 16 + nc;
      if (TRANS) {          // Vt[o][n]: 8 consecutive rows -> one 16B store
        v8bf pk;
#pragma unroll
        for (int r = 0; r < 8; ++r) pk[r] = (bf16_t)acc[i][j][r];
        *(v8bf*)(Y + (size_t)ncol * nrows + mrow) = pk;
      } else {
#pragma unroll
        for (int r = 0; r < 8; ++r)
          Y[(size_t)(mrow + r) * ocols + ncol] = (bf16_t)acc[i][j][r];
      }
    }
}

// ---- kernel 2: P_unnorm = exp(QK^T*scale + causal_mask); l += rowsum -----
__global__ __launch_bounds__(256) void attn_scores_kernel(
    const bf16_t* __restrict__ Q, const bf16_t* __restrict__ K,
    bf16_t* __restrict__ P, float* __restrict__ lsum, int dmodel, int n)
{
  __shared__ alignas(16) bf16_t As0[TILE_M * LDT];
  __shared__ alignas(16) bf16_t Bs0[TILE_N * LDT];
  __shared__ alignas(16) bf16_t As1[TILE_M * LDT];
  __shared__ alignas(16) bf16_t Bs1[TILE_N * LDT];
  GEMM_PREAMBLE();

  bf16_gemm_loop((const unsigned int*)Q, (const unsigned int*)K,
                 As0, Bs0, As1, Bs1, m0, n0, dmodel >> 1, dmodel / TILE_K,
                 wm, wn, lane, tid, acc);

  const float scale = 0.03125f;  // 1/sqrt(1024)
  const int nc = lane & 15, mb = (lane >> 4) * 8;
#pragma unroll
  for (int i = 0; i < 4; ++i) {
#pragma unroll
    for (int r = 0; r < 8; ++r) {
      const int m = m0 + wm + i * 16 + mb + r;
      float rowsum = 0.0f;
#pragma unroll
      for (int j = 0; j < 2; ++j) {
        const int col = n0 + wn + j * 16 + nc;
        float s = acc[i][j][r] * scale + ((col > m) ? -10000.0f : 0.0f);
        float p = __expf(s);          // masked entries underflow to exactly 0
        P[(size_t)m * n + col] = (bf16_t)p;
        rowsum += p;
      }
      rowsum += __shfl_xor(rowsum, 1);
      rowsum += __shfl_xor(rowsum, 2);
      rowsum += __shfl_xor(rowsum, 4);
      rowsum += __shfl_xor(rowsum, 8);
      if (nc == 0) atomicAdd(&lsum[m], rowsum);
    }
  }
}

// ---- kernel 3: O = (P_unnorm @ V) / l  (V held transposed: Vt[o][n]) -----
__global__ __launch_bounds__(256) void attn_out_kernel(
    const bf16_t* __restrict__ P, const bf16_t* __restrict__ Vt,
    const float* __restrict__ lsum, float* __restrict__ O, int n, int dout)
{
  __shared__ alignas(16) bf16_t As0[TILE_M * LDT];
  __shared__ alignas(16) bf16_t Bs0[TILE_N * LDT];
  __shared__ alignas(16) bf16_t As1[TILE_M * LDT];
  __shared__ alignas(16) bf16_t Bs1[TILE_N * LDT];
  GEMM_PREAMBLE();

  bf16_gemm_loop((const unsigned int*)P, (const unsigned int*)Vt,
                 As0, Bs0, As1, Bs1, m0, n0, n >> 1, n / TILE_K,
                 wm, wn, lane, tid, acc);

  const int nc = lane & 15, mb = (lane >> 4) * 8;
#pragma unroll
  for (int i = 0; i < 4; ++i)
#pragma unroll
    for (int r = 0; r < 8; ++r) {
      const int m = m0 + wm + i * 16 + mb + r;
      const float inv_l = 1.0f / lsum[m];
#pragma unroll
      for (int j = 0; j < 2; ++j) {
        const int col = n0 + wn + j * 16 + nc;
        O[(size_t)m * dout + col] = acc[i][j][r] * inv_l;
      }
    }
}

extern "C" void kernel_launch(void* const* d_in, const int* in_sizes, int n_in,
                              void* d_out, int out_size, void* d_ws, size_t ws_size,
                              hipStream_t stream) {
  (void)in_sizes; (void)n_in; (void)out_size; (void)ws_size;
  const int N = 4096, DIN = 1024, DOUT = 1024;

  const float* qx = (const float*)d_in[0];
  const float* kx = (const float*)d_in[1];
  const float* vx = (const float*)d_in[2];
  const float* Wq = (const float*)d_in[3];
  const float* Wk = (const float*)d_in[4];
  const float* Wv = (const float*)d_in[5];
  float* out = (float*)d_out;

  char* ws = (char*)d_ws;
  bf16_t* Qb = (bf16_t*)(ws);                                  //  8 MB  [N, DOUT]
  bf16_t* Kb = (bf16_t*)(ws + ((size_t)8 << 20));              //  8 MB  [N, DOUT]
  bf16_t* Vt = (bf16_t*)(ws + ((size_t)16 << 20));             //  8 MB  [DOUT, N]
  float*  l  = (float*) (ws + ((size_t)24 << 20));             // 16 KB  [N]
  bf16_t* P  = (bf16_t*)(ws + ((size_t)24 << 20) + (1 << 16)); // 32 MB  [N, N]

  dim3 blk(256);
  init_zero_kernel<<<dim3(N / 256), blk, 0, stream>>>(l);

  qkv_proj_kernel<0><<<dim3(DOUT / TILE_N, N / TILE_M), blk, 0, stream>>>(
      qx, Wq, Qb, DIN, N, DOUT);
  qkv_proj_kernel<0><<<dim3(DOUT / TILE_N, N / TILE_M), blk, 0, stream>>>(
      kx, Wk, Kb, DIN, N, DOUT);
  qkv_proj_kernel<1><<<dim3(DOUT / TILE_N, N / TILE_M), blk, 0, stream>>>(
      vx, Wv, Vt, DIN, N, DOUT);

  attn_scores_kernel<<<dim3(N / TILE_N, N / TILE_M), blk, 0, stream>>>(
      Qb, Kb, P, l, DOUT, N);

  attn_out_kernel<<<dim3(DOUT / TILE_N, N / TILE_M), blk, 0, stream>>>(
      P, Vt, l, out, N, DOUT);
}
